// APPNP_85650237816961
// MI455X (gfx1250) — matrix operand order repaired and checked by reading, not compiled, not run on hardware
//
#include <hip/hip_runtime.h>

// ---------------------------------------------------------------------------
// APPNP on MI455X (gfx1250):
//   MLP (relu(x@Win^T+bin))@Wout^T+bout  -> bf16 WMMA (v_wmma_f32_16x16x32_bf16)
//   10x propagation: znext = (1-a)*z0 ; znext[dst] += a*w*zcur[src] (f32 atomics,
//   z (25.6MB) + edges (64MB) are L2-resident on the 192MB L2).
// ---------------------------------------------------------------------------

#define ALPHA_F 0.9f

typedef __attribute__((ext_vector_type(16))) __bf16 v16bf;
typedef __attribute__((ext_vector_type(8)))  __bf16 v8bf;
typedef __attribute__((ext_vector_type(8)))  float  v8f;

__device__ __forceinline__ unsigned short f2bf(float f) {
  union { float f; unsigned u; } v; v.f = f;
  unsigned r = v.u + 0x7FFFu + ((v.u >> 16) & 1u);   // round-to-nearest-even
  return (unsigned short)(r >> 16);
}

__device__ __forceinline__ v16bf cat8(v8bf lo, v8bf hi) {
  return __builtin_shufflevector(lo, hi, 0,1,2,3,4,5,6,7,8,9,10,11,12,13,14,15);
}

// LDS row strides (in bf16 elements); padding spreads LDS banks.
#define SA_STRIDE 520   // 512 + 8
#define SH_STRIDE 264   // 256 + 8
#define SB_STRIDE 40    //  32 + 8

// A-fragment (16x32 bf16, MxK): lane holds row m = lane%16;
//   VGPR0-3 hold K {0..7 | 8..15}, VGPR4-7 hold K {16..23 | 24..31} by lane half.
__device__ __forceinline__ v16bf load_a_frag(const unsigned short* rowp, int lane) {
  const __bf16* ap = (const __bf16*)rowp;
  int g0 = (lane < 16) ? 0 : 8;
  return cat8(*(const v8bf*)(ap + g0), *(const v8bf*)(ap + g0 + 16));
}

// B-fragment (32x16 bf16, KxN): lane holds column n = lane%16; K {0..7 | 8..15}, {16..23 | 24..31} by lane half.
__device__ __forceinline__ v16bf load_b_frag(const unsigned short* rowp, int lane) {
  const __bf16* bp = (const __bf16*)rowp;
  int g0 = (lane < 16) ? 0 : 8;
  v16bf b;
  #pragma unroll
  for (int s = 0; s < 8; ++s) {
    b[s]     = bp[(g0 + s) * SB_STRIDE];
    b[s + 8] = bp[(g0 + 16 + s) * SB_STRIDE];
  }
  return b;
}

__global__ __launch_bounds__(256, 1)
void appnp_mlp_wmma(const float* __restrict__ x,
                    const float* __restrict__ Win,
                    const float* __restrict__ bin,
                    const float* __restrict__ Wout,
                    const float* __restrict__ bout,
                    float* __restrict__ z0,
                    int n_nodes)
{
  __shared__ __align__(16) unsigned short sA[64 * SA_STRIDE];  //  66.5 KB
  __shared__ __align__(16) unsigned short sH[64 * SH_STRIDE];  //  33.8 KB
  __shared__ __align__(16) unsigned short sB[512 * SB_STRIDE]; //  40.0 KB

  const int t        = threadIdx.x;
  const int lane     = t & 31;
  const int wid      = t >> 5;
  const int strip    = wid & 3;   // which 16-row strip of the 64-row tile
  const int nsel     = wid >> 2;  // which 16-col half of the staged 32 cols
  const int nodebase = blockIdx.x * 64;
  const int mrow     = strip * 16 + (lane & 15);

  // ---- stage A tile: x[nodebase..+63][0..511] fp32 -> bf16 LDS ----
  #pragma unroll 4
  for (int i = 0; i < 32; ++i) {
    int fid  = i * 256 + t;        // 0..8191 float4 slots
    int row  = fid >> 7;           // /128
    int quad = fid & 127;
    float4 v = make_float4(0.f, 0.f, 0.f, 0.f);
    int node = nodebase + row;
    if (node < n_nodes) v = ((const float4*)x)[node * 128 + quad];
    unsigned u0 = (unsigned)f2bf(v.x) | ((unsigned)f2bf(v.y) << 16);
    unsigned u1 = (unsigned)f2bf(v.z) | ((unsigned)f2bf(v.w) << 16);
    unsigned* p = (unsigned*)&sA[row * SA_STRIDE + quad * 4];
    p[0] = u0; p[1] = u1;
  }
  __syncthreads();

  // ---- GEMM1: h = relu(x @ Win^T + bin), K=512, N=256 in 8 groups of 32 ----
  for (int np = 0; np < 8; ++np) {
    const int nbase = np * 32;
    { // stage sB[k][nl] = Win[nbase+nl][k]  (bf16, transposed)
      int nl = t & 31;
      int kb = (t >> 5) * 64;
      const float4* wrow = (const float4*)(Win + (size_t)(nbase + nl) * 512);
      #pragma unroll 4
      for (int kk = 0; kk < 64; kk += 4) {
        float4 w = wrow[(kb + kk) >> 2];
        sB[(kb + kk + 0) * SB_STRIDE + nl] = f2bf(w.x);
        sB[(kb + kk + 1) * SB_STRIDE + nl] = f2bf(w.y);
        sB[(kb + kk + 2) * SB_STRIDE + nl] = f2bf(w.z);
        sB[(kb + kk + 3) * SB_STRIDE + nl] = f2bf(w.w);
      }
    }
    __syncthreads();

    v8f acc = {0.f,0.f,0.f,0.f,0.f,0.f,0.f,0.f};
    const unsigned short* arow = &sA[mrow * SA_STRIDE];
    #pragma unroll
    for (int kc = 0; kc < 16; ++kc) {
      v16bf aF = load_a_frag(arow + kc * 32, lane);
      v16bf bF = load_b_frag(&sB[(kc * 32) * SB_STRIDE + nsel * 16 + (lane & 15)], lane);
      acc = __builtin_amdgcn_wmma_f32_16x16x32_bf16(
          false, aF, false, bF, (short)0, acc, false, false);
    }

    const int ncol = nbase + nsel * 16 + (lane & 15);
    const float bias = bin[ncol];
    #pragma unroll
    for (int r = 0; r < 8; ++r) {
      int ml = strip * 16 + ((lane < 16) ? r : r + 8);
      float v = acc[r] + bias;
      v = v > 0.f ? v : 0.f;                 // relu
      sH[ml * SH_STRIDE + ncol] = f2bf(v);
    }
    __syncthreads();
  }

  // ---- GEMM2: z0 = h @ Wout^T + bout, K=256, N=64 in 2 groups of 32 ----
  for (int np = 0; np < 2; ++np) {
    const int nbase = np * 32;
    { // stage sB[k][nl] = Wout[nbase+nl][k]
      int nl = t & 31;
      int kb = (t >> 5) * 32;
      const float4* wrow = (const float4*)(Wout + (size_t)(nbase + nl) * 256);
      #pragma unroll 4
      for (int kk = 0; kk < 32; kk += 4) {
        float4 w = wrow[(kb + kk) >> 2];
        sB[(kb + kk + 0) * SB_STRIDE + nl] = f2bf(w.x);
        sB[(kb + kk + 1) * SB_STRIDE + nl] = f2bf(w.y);
        sB[(kb + kk + 2) * SB_STRIDE + nl] = f2bf(w.z);
        sB[(kb + kk + 3) * SB_STRIDE + nl] = f2bf(w.w);
      }
    }
    __syncthreads();

    v8f acc = {0.f,0.f,0.f,0.f,0.f,0.f,0.f,0.f};
    const unsigned short* arow = &sH[mrow * SH_STRIDE];
    #pragma unroll
    for (int kc = 0; kc < 8; ++kc) {
      v16bf aF = load_a_frag(arow + kc * 32, lane);
      v16bf bF = load_b_frag(&sB[(kc * 32) * SB_STRIDE + nsel * 16 + (lane & 15)], lane);
      acc = __builtin_amdgcn_wmma_f32_16x16x32_bf16(
          false, aF, false, bF, (short)0, acc, false, false);
    }

    const int c = nbase + nsel * 16 + (lane & 15);
    const float bias = bout[c];
    #pragma unroll
    for (int r = 0; r < 8; ++r) {
      int node = nodebase + strip * 16 + ((lane < 16) ? r : r + 8);
      if (node < n_nodes) z0[(size_t)node * 64 + c] = acc[r] + bias;
    }
    __syncthreads();
  }
}

// znext = (1-alpha) * z0   (fully rewrites the buffer each layer)
__global__ __launch_bounds__(256)
void appnp_init(const float* __restrict__ z0, float* __restrict__ zn, int n) {
  int i = blockIdx.x * 256 + threadIdx.x;
  if (i < n) zn[i] = (1.0f - ALPHA_F) * z0[i];
}

// znext[dst] += alpha * w * zcur[src] ; one thread handles 4 channels of an edge.
__global__ __launch_bounds__(256)
void appnp_scatter(const int* __restrict__ ei,
                   const float* __restrict__ ew,
                   const float* __restrict__ zc,
                   float* __restrict__ zn,
                   int E) {
  int idx = blockIdx.x * 256 + threadIdx.x;
  int e = idx >> 4;
  if (e >= E) return;
  int q   = (idx & 15) * 4;
  int src = ei[e];
  int dst = ei[E + e];
  float w = ALPHA_F * ew[e];
  const float4 zv = *(const float4*)(zc + (size_t)src * 64 + q);
  float* o = zn + (size_t)dst * 64 + q;
  unsafeAtomicAdd(o + 0, w * zv.x);
  unsafeAtomicAdd(o + 1, w * zv.y);
  unsafeAtomicAdd(o + 2, w * zv.z);
  unsafeAtomicAdd(o + 3, w * zv.w);
}

extern "C" void kernel_launch(void* const* d_in, const int* in_sizes, int n_in,
                              void* d_out, int out_size, void* d_ws, size_t ws_size,
                              hipStream_t stream) {
  const float* x    = (const float*)d_in[0];
  const float* Win  = (const float*)d_in[1];
  const float* bin  = (const float*)d_in[2];
  const float* Wout = (const float*)d_in[3];
  const float* bout = (const float*)d_in[4];
  const int*   ei   = (const int*)d_in[5];
  const float* ew   = (const float*)d_in[6];

  const int n_nodes = in_sizes[0] / 512;
  const int E       = in_sizes[6];
  const int nz      = n_nodes * 64;

  float* z0   = (float*)d_ws;        // [n_nodes*64]
  float* zA   = z0 + nz;             // ping buffer
  float* zOut = (float*)d_out;       // doubles as the odd-iteration buffer

  dim3 blk(256);
  appnp_mlp_wmma<<<dim3((n_nodes + 63) / 64), blk, 0, stream>>>(
      x, Win, bin, Wout, bout, z0, n_nodes);

  const float* cur = z0;
  for (int it = 0; it < 10; ++it) {
    float* nxt = (it & 1) ? zOut : zA;          // it=9 (last) lands in d_out
    appnp_init<<<dim3((nz + 255) / 256), blk, 0, stream>>>(z0, nxt, nz);
    appnp_scatter<<<dim3(((size_t)E * 16 + 255) / 256), blk, 0, stream>>>(
        ei, ew, cur, nxt, E);
    cur = nxt;
  }
}
